// StreamingSNNSentinel_89593017794587
// MI455X (gfx1250) — matrix operand order repaired and checked by reading, not compiled, not run on hardware
//
#include <hip/hip_runtime.h>
#include <hip/hip_bf16.h>

typedef __attribute__((ext_vector_type(16))) __bf16 v16bf;
typedef __attribute__((ext_vector_type(8)))  __bf16 v8bf;
typedef __attribute__((ext_vector_type(8)))  float  v8f;
typedef __attribute__((ext_vector_type(4)))  float  v4f;

namespace {
constexpr int kB = 512, kT = 2048, kD = 64, kH = 48, kC = 3;
constexpr int TB = 16;              // batch-rows per tile (WMMA M)
constexpr int KP = 64;              // padded K for LDS tiles
constexpr int kL = 64;              // chunk length (time steps)
constexpr int kNCH = kT / kL;       // 32 chunks
constexpr int kNT = kB / TB;        // 32 batch tiles
constexpr float kBeta = 0.9f;
constexpr size_t kPart = (size_t)kNT * kNCH * TB * kH;   // per-ws-buffer floats
}

__device__ __forceinline__ v8f wmma_bf16(v16bf a, v16bf b, v8f c) {
  return __builtin_amdgcn_wmma_f32_16x16x32_bf16(false, a, false, b, (short)0, c,
                                                 false, false);
}

__device__ __forceinline__ void lds_fence() {
  __asm__ volatile("s_wait_dscnt 0" ::: "memory");
  __builtin_amdgcn_wave_barrier();
}

// fast sigmoid: v_exp_f32 + v_rcp_f32 (avoid IEEE div expansion)
__device__ __forceinline__ float fast_sigmoid(float v) {
  return __builtin_amdgcn_rcpf(1.f + __expf(-v));
}

// B-fragment (32x16 bf16) from LDS matrix [row=N][col=K(pad 64)].
__device__ __forceinline__ v16bf ld_bfrag(const __bf16* base, int row0, int lane, int kt) {
  const int n  = lane & 15;
  const int kh = (lane >> 4) << 4;
  const __bf16* p = base + (row0 + n) * KP + kt * 32 + kh;
  v8bf lo = *(const v8bf*)p;
  v8bf hi = *(const v8bf*)(p + 8);
  v16bf r;
#pragma unroll
  for (int i = 0; i < 8; ++i) { r[i] = lo[i]; r[i + 8] = hi[i]; }
  return r;
}

// A-fragment (16x32 bf16) from LDS tile [m][k(pad 64)].
__device__ __forceinline__ v16bf ld_afrag_lds(const __bf16* base, int lane, int kt) {
  const int m  = lane & 15;
  const int k0 = kt * 32 + ((lane >> 4) << 4);
  const __bf16* p = base + m * KP + k0;
  v8bf c0 = *(const v8bf*)p;
  v8bf c1 = *(const v8bf*)(p + 8);
  v16bf r;
#pragma unroll
  for (int i = 0; i < 8; ++i) { r[i] = c0[i]; r[i + 8] = c1[i]; }
  return r;
}

// A-fragment from a global f32 row (streamed once -> non-temporal loads).
__device__ __forceinline__ v16bf ld_afrag_f32(const float* row, int lane, int kt) {
  const int k0 = kt * 32 + ((lane >> 4) << 4);
  v4f a0 = __builtin_nontemporal_load((const v4f*)(row + k0));
  v4f a1 = __builtin_nontemporal_load((const v4f*)(row + k0 + 4));
  v4f b0 = __builtin_nontemporal_load((const v4f*)(row + k0 + 8));
  v4f b1 = __builtin_nontemporal_load((const v4f*)(row + k0 + 12));
  v16bf r;
#pragma unroll
  for (int i = 0; i < 4; ++i) {
    r[i]      = (__bf16)a0[i];
    r[i + 4]  = (__bf16)a1[i];
    r[i + 8]  = (__bf16)b0[i];
    r[i + 12] = (__bf16)b1[i];
  }
  return r;
}

// ---------------------------------------------------------------------------
// Phases A (EMIT=false) and C (EMIT=true): per-(tile,chunk) recurrence using
// folded weights: mem_t = mem_{t-1} @ Afold + (1-b)*(x_t@Wenc^T) + bias_c,
// Afold = b*I + (1-b)*Wmem^T. Phase A writes chunk endpoints; phase C seeds
// from Minit and emits z, recon, zsum partials.
// ---------------------------------------------------------------------------
template <bool EMIT>
__global__ __launch_bounds__(32)
void snn_chunk_kernel(const float* __restrict__ x,
                      const float* __restrict__ Wenc, const float* __restrict__ benc,
                      const float* __restrict__ Wmem, const float* __restrict__ bmem,
                      const float* __restrict__ Wdec, const float* __restrict__ bdec,
                      const float* __restrict__ Minit, float* __restrict__ Eout,
                      float* __restrict__ recon, float* __restrict__ zout,
                      float* __restrict__ Zpart) {
  __shared__ __bf16 lds_wenc[kH * KP];
  __shared__ __bf16 lds_wdec[kD * KP];
  __shared__ __bf16 lds_mem[TB * KP];
  __shared__ __bf16 lds_z[TB * KP];

  const int lane  = threadIdx.x;
  const int chunk = blockIdx.x;
  const int tile  = blockIdx.y;
  const int b0    = tile * TB;
  const int t0    = chunk * kL;
  const int n     = lane & 15;
  const int mhi   = (lane >> 4) << 3;
  const float om  = 1.f - kBeta;

  // stage enc weights pre-scaled by (1-beta); dec weights (phase C only)
  for (int i = lane; i < kH * KP; i += 32) lds_wenc[i] = (__bf16)(om * Wenc[i]);
  if (EMIT) {
    for (int i = lane; i < kD * KP; i += 32) {
      const int k = i & (KP - 1);
      lds_wdec[i] = (k < kH) ? (__bf16)Wdec[(i >> 6) * kH + k] : (__bf16)0.f;
    }
  }
  for (int i = lane; i < TB * KP; i += 32) {
    lds_mem[i] = (__bf16)0.f;
    lds_z[i]   = (__bf16)0.f;
  }

  // Afold B-fragments in registers: B[k][h] = beta*(k==h) + (1-beta)*Wmem[h][k]
  v16bf bAf[2][3];
#pragma unroll
  for (int kt = 0; kt < 2; ++kt) {
#pragma unroll
    for (int nt = 0; nt < 3; ++nt) {
      const int row = nt * 16 + n;
      const int k0  = kt * 32 + ((lane >> 4) << 4);
#pragma unroll
      for (int i = 0; i < 16; ++i) {
        const int k = k0 + i;
        float v = 0.f;
        if (k < kH) v = om * Wmem[row * kH + k] + ((k == row) ? kBeta : 0.f);
        bAf[kt][nt][i] = (__bf16)v;
      }
    }
  }

  float bias_c[3], bias_d[4];
#pragma unroll
  for (int nt = 0; nt < 3; ++nt) bias_c[nt] = om * (benc[nt * 16 + n] + bmem[nt * 16 + n]);
#pragma unroll
  for (int j = 0; j < 4; ++j) bias_d[j] = EMIT ? bdec[j * 16 + n] : 0.f;

  float zsum[3][8];
#pragma unroll
  for (int nt = 0; nt < 3; ++nt)
#pragma unroll
    for (int r = 0; r < 8; ++r) zsum[nt][r] = 0.f;

  // Phase C: seed mem state from Minit
  if (EMIT) {
    const float* Mi = Minit + ((size_t)(tile * kNCH + chunk)) * TB * kH;
#pragma unroll
    for (int nt = 0; nt < 3; ++nt) {
      const int h = nt * 16 + n;
#pragma unroll
      for (int r = 0; r < 8; ++r) {
        const int m = r + mhi;
        lds_mem[m * KP + h] = (__bf16)Mi[m * kH + h];
      }
    }
  }
  __syncthreads();

  const float* xrow = x + ((size_t)(b0 + n) * kT + t0) * kD;
  v8f mstate[3];

#pragma unroll 1
  for (int tt = 0; tt < kL; ++tt) {
    const int t = t0 + tt;

    v16bf ax0 = ld_afrag_f32(xrow, lane, 0);
    v16bf ax1 = ld_afrag_f32(xrow, lane, 1);
    __builtin_prefetch(xrow + 2 * kD, 0, 0);

    v16bf am0 = ld_afrag_lds(lds_mem, lane, 0);
    v16bf am1 = ld_afrag_lds(lds_mem, lane, 1);

    // mem_new directly from the accumulator (beta folded into weights)
#pragma unroll
    for (int nt = 0; nt < 3; ++nt) {
      const float bb = bias_c[nt];
      v8f acc = {bb, bb, bb, bb, bb, bb, bb, bb};
      acc = wmma_bf16(ax0, ld_bfrag(lds_wenc, nt * 16, lane, 0), acc);
      acc = wmma_bf16(ax1, ld_bfrag(lds_wenc, nt * 16, lane, 1), acc);
      acc = wmma_bf16(am0, bAf[0][nt], acc);
      acc = wmma_bf16(am1, bAf[1][nt], acc);
      mstate[nt] = acc;
    }

    // relayout mem (and z for phase C) into LDS; stream z out
#pragma unroll
    for (int nt = 0; nt < 3; ++nt) {
      const int h = nt * 16 + n;
#pragma unroll
      for (int r = 0; r < 8; ++r) {
        const int m = r + mhi;
        const float mn = mstate[nt][r];
        lds_mem[m * KP + h] = (__bf16)mn;
        if (EMIT) {
          const float zz = fast_sigmoid(mn);
          zsum[nt][r] += zz;
          lds_z[m * KP + h] = (__bf16)zz;
          __builtin_nontemporal_store(zz, zout + ((size_t)(b0 + m) * kT + t) * kH + h);
        }
      }
    }
    lds_fence();

    if (EMIT) {
      v16bf az0 = ld_afrag_lds(lds_z, lane, 0);
      v16bf az1 = ld_afrag_lds(lds_z, lane, 1);
#pragma unroll
      for (int j = 0; j < 4; ++j) {
        const float bb = bias_d[j];
        v8f acc = {bb, bb, bb, bb, bb, bb, bb, bb};
        acc = wmma_bf16(az0, ld_bfrag(lds_wdec, j * 16, lane, 0), acc);
        acc = wmma_bf16(az1, ld_bfrag(lds_wdec, j * 16, lane, 1), acc);
        const int dcol = j * 16 + n;
#pragma unroll
        for (int r = 0; r < 8; ++r) {
          const int m = r + mhi;
          __builtin_nontemporal_store(acc[r],
              recon + ((size_t)(b0 + m) * kT + t) * kD + dcol);
        }
      }
    }
    xrow += kD;
  }

  if (EMIT) {
    float* Zp = Zpart + ((size_t)(tile * kNCH + chunk)) * TB * kH;
#pragma unroll
    for (int nt = 0; nt < 3; ++nt)
#pragma unroll
      for (int r = 0; r < 8; ++r)
        Zp[(r + mhi) * kH + nt * 16 + n] = zsum[nt][r];
  } else {
    float* Ep = Eout + ((size_t)(tile * kNCH + chunk)) * TB * kH;
#pragma unroll
    for (int nt = 0; nt < 3; ++nt)
#pragma unroll
      for (int r = 0; r < 8; ++r)
        Ep[(r + mhi) * kH + nt * 16 + n] = mstate[nt][r];
  }
}

// ---------------------------------------------------------------------------
// Phase B: A^64 via 6 LDS squarings, then 32-step scan over chunk endpoints.
// One block per batch tile (f32 throughout; trivial FLOPs).
// ---------------------------------------------------------------------------
__global__ __launch_bounds__(32)
void snn_scanfix_kernel(const float* __restrict__ Wmem,
                        const float* __restrict__ E,
                        float* __restrict__ Minit) {
  __shared__ float Abuf[2][kH * kH];
  __shared__ float St[2][TB * kH];
  const int lane = threadIdx.x;
  const int tile = blockIdx.x;
  const float om = 1.f - kBeta;

  for (int i = lane; i < kH * kH; i += 32) {
    const int k = i / kH, h = i % kH;
    Abuf[0][i] = ((k == h) ? kBeta : 0.f) + om * Wmem[h * kH + k];
  }
  __syncthreads();

  int src = 0;
  for (int s = 0; s < 6; ++s) {      // -> A^64
    for (int i = lane; i < kH * kH; i += 32) {
      const int r = i / kH, c = i % kH;
      float acc = 0.f;
      for (int k = 0; k < kH; ++k) acc += Abuf[src][r * kH + k] * Abuf[src][k * kH + c];
      Abuf[src ^ 1][i] = acc;
    }
    __syncthreads();
    src ^= 1;
  }

  for (int i = lane; i < TB * kH; i += 32) St[0][i] = 0.f;
  __syncthreads();
  int ss = 0;
  for (int j = 0; j < kNCH; ++j) {
    const float* Ej = E + ((size_t)(tile * kNCH + j)) * TB * kH;
    float* Mj = Minit + ((size_t)(tile * kNCH + j)) * TB * kH;
    for (int i = lane; i < TB * kH; i += 32) Mj[i] = St[ss][i];   // state BEFORE chunk j
    for (int i = lane; i < TB * kH; i += 32) {
      const int m = i / kH, h = i % kH;
      float acc = Ej[i];
      for (int k = 0; k < kH; ++k) acc += St[ss][m * kH + k] * Abuf[src][k * kH + h];
      St[ss ^ 1][i] = acc;
    }
    __syncthreads();
    ss ^= 1;
  }
}

// ---------------------------------------------------------------------------
// Phase D: deterministic (chunk-ordered) zsum reduction + logits.
// ---------------------------------------------------------------------------
__global__ __launch_bounds__(32)
void snn_logits_kernel(const float* __restrict__ Zpart,
                       const float* __restrict__ Wcls, const float* __restrict__ bcls,
                       float* __restrict__ logits) {
  __shared__ float zacc[TB * kH];
  const int lane = threadIdx.x;
  const int tile = blockIdx.x;
  for (int i = lane; i < TB * kH; i += 32) {
    float a = 0.f;
    for (int j = 0; j < kNCH; ++j)
      a += Zpart[((size_t)(tile * kNCH + j)) * TB * kH + i];
    zacc[i] = a;
  }
  __syncthreads();
  if (lane < 16) {
    const int m = lane;
    float a0 = bcls[0], a1 = bcls[1], a2 = bcls[2];
    const float inv = 1.f / (float)kT;
    for (int h = 0; h < kH; ++h) {
      const float zm = zacc[m * kH + h] * inv;
      a0 += zm * Wcls[0 * kH + h];
      a1 += zm * Wcls[1 * kH + h];
      a2 += zm * Wcls[2 * kH + h];
    }
    float* lp = logits + (size_t)(tile * TB + m) * kC;
    lp[0] = a0; lp[1] = a1; lp[2] = a2;
  }
}

// ---------------------------------------------------------------------------
// Fallback: fully-fused single-pass kernel (used if ws too small).
// ---------------------------------------------------------------------------
__global__ __launch_bounds__(32)
void snn_fused_kernel(const float* __restrict__ x,
                      const float* __restrict__ Wenc, const float* __restrict__ benc,
                      const float* __restrict__ Wmem, const float* __restrict__ bmem,
                      const float* __restrict__ Wdec, const float* __restrict__ bdec,
                      const float* __restrict__ Wcls, const float* __restrict__ bcls,
                      float* __restrict__ recon, float* __restrict__ logits,
                      float* __restrict__ zout) {
  __shared__ __bf16 lds_wenc[kH * KP];
  __shared__ __bf16 lds_wdec[kD * KP];
  __shared__ __bf16 lds_mem[TB * KP];
  __shared__ __bf16 lds_z[TB * KP];
  __shared__ float  lds_zsum[TB * kH];

  const int lane = threadIdx.x;
  const int b0   = blockIdx.x * TB;
  const int n    = lane & 15;
  const int mhi  = (lane >> 4) << 3;
  const float om = 1.f - kBeta;

  for (int i = lane; i < kH * KP; i += 32) lds_wenc[i] = (__bf16)(om * Wenc[i]);
  for (int i = lane; i < kD * KP; i += 32) {
    const int k = i & (KP - 1);
    lds_wdec[i] = (k < kH) ? (__bf16)Wdec[(i >> 6) * kH + k] : (__bf16)0.f;
  }
  for (int i = lane; i < TB * KP; i += 32) {
    lds_mem[i] = (__bf16)0.f;
    lds_z[i]   = (__bf16)0.f;
  }

  v16bf bAf[2][3];
#pragma unroll
  for (int kt = 0; kt < 2; ++kt)
#pragma unroll
    for (int nt = 0; nt < 3; ++nt) {
      const int row = nt * 16 + n;
      const int k0  = kt * 32 + ((lane >> 4) << 4);
#pragma unroll
      for (int i = 0; i < 16; ++i) {
        const int k = k0 + i;
        float v = 0.f;
        if (k < kH) v = om * Wmem[row * kH + k] + ((k == row) ? kBeta : 0.f);
        bAf[kt][nt][i] = (__bf16)v;
      }
    }

  float bias_c[3], bias_d[4];
#pragma unroll
  for (int nt = 0; nt < 3; ++nt) bias_c[nt] = om * (benc[nt * 16 + n] + bmem[nt * 16 + n]);
#pragma unroll
  for (int j = 0; j < 4; ++j) bias_d[j] = bdec[j * 16 + n];

  float zsum[3][8];
#pragma unroll
  for (int nt = 0; nt < 3; ++nt)
#pragma unroll
    for (int r = 0; r < 8; ++r) zsum[nt][r] = 0.f;

  __syncthreads();
  const float* xrow = x + (size_t)(b0 + n) * kT * kD;

#pragma unroll 1
  for (int t = 0; t < kT; ++t) {
    v16bf ax0 = ld_afrag_f32(xrow, lane, 0);
    v16bf ax1 = ld_afrag_f32(xrow, lane, 1);
    __builtin_prefetch(xrow + 2 * kD, 0, 0);
    v16bf am0 = ld_afrag_lds(lds_mem, lane, 0);
    v16bf am1 = ld_afrag_lds(lds_mem, lane, 1);

    v8f mstate[3];
#pragma unroll
    for (int nt = 0; nt < 3; ++nt) {
      const float bb = bias_c[nt];
      v8f acc = {bb, bb, bb, bb, bb, bb, bb, bb};
      acc = wmma_bf16(ax0, ld_bfrag(lds_wenc, nt * 16, lane, 0), acc);
      acc = wmma_bf16(ax1, ld_bfrag(lds_wenc, nt * 16, lane, 1), acc);
      acc = wmma_bf16(am0, bAf[0][nt], acc);
      acc = wmma_bf16(am1, bAf[1][nt], acc);
      mstate[nt] = acc;
    }

#pragma unroll
    for (int nt = 0; nt < 3; ++nt) {
      const int h = nt * 16 + n;
#pragma unroll
      for (int r = 0; r < 8; ++r) {
        const int m = r + mhi;
        const float mn = mstate[nt][r];
        const float zz = fast_sigmoid(mn);
        zsum[nt][r] += zz;
        lds_mem[m * KP + h] = (__bf16)mn;
        lds_z[m * KP + h]   = (__bf16)zz;
        __builtin_nontemporal_store(zz, zout + ((size_t)(b0 + m) * kT + t) * kH + h);
      }
    }
    lds_fence();

    v16bf az0 = ld_afrag_lds(lds_z, lane, 0);
    v16bf az1 = ld_afrag_lds(lds_z, lane, 1);
#pragma unroll
    for (int j = 0; j < 4; ++j) {
      const float bb = bias_d[j];
      v8f acc = {bb, bb, bb, bb, bb, bb, bb, bb};
      acc = wmma_bf16(az0, ld_bfrag(lds_wdec, j * 16, lane, 0), acc);
      acc = wmma_bf16(az1, ld_bfrag(lds_wdec, j * 16, lane, 1), acc);
      const int dcol = j * 16 + n;
#pragma unroll
      for (int r = 0; r < 8; ++r) {
        const int m = r + mhi;
        __builtin_nontemporal_store(acc[r],
            recon + ((size_t)(b0 + m) * kT + t) * kD + dcol);
      }
    }
    xrow += kD;
  }

#pragma unroll
  for (int nt = 0; nt < 3; ++nt)
#pragma unroll
    for (int r = 0; r < 8; ++r)
      lds_zsum[(r + mhi) * kH + nt * 16 + n] = zsum[nt][r];
  __syncthreads();
  if (lane < 16) {
    const int m = lane;
    float a0 = bcls[0], a1 = bcls[1], a2 = bcls[2];
    const float inv = 1.f / (float)kT;
    for (int h = 0; h < kH; ++h) {
      const float zm = lds_zsum[m * kH + h] * inv;
      a0 += zm * Wcls[0 * kH + h];
      a1 += zm * Wcls[1 * kH + h];
      a2 += zm * Wcls[2 * kH + h];
    }
    float* lp = logits + (size_t)(b0 + m) * kC;
    lp[0] = a0; lp[1] = a1; lp[2] = a2;
  }
}

extern "C" void kernel_launch(void* const* d_in, const int* in_sizes, int n_in,
                              void* d_out, int out_size, void* d_ws, size_t ws_size,
                              hipStream_t stream) {
  const float* x    = (const float*)d_in[0];
  const float* Wenc = (const float*)d_in[1];
  const float* benc = (const float*)d_in[2];
  const float* Wmem = (const float*)d_in[3];
  const float* bmem = (const float*)d_in[4];
  const float* Wdec = (const float*)d_in[5];
  const float* bdec = (const float*)d_in[6];
  const float* Wcls = (const float*)d_in[7];
  const float* bcls = (const float*)d_in[8];

  float* recon  = (float*)d_out;                 // [B,T,D]
  float* logits = recon + (size_t)kB * kT * kD;  // [B,C]
  float* zout   = logits + (size_t)kB * kC;      // [B,T,H]

  const size_t need = 3 * kPart * sizeof(float);
  if (ws_size >= need) {
    float* E     = (float*)d_ws;
    float* Minit = E + kPart;
    float* Zpart = Minit + kPart;

    // Phase A: chunk endpoints (zero-init), 1024-way parallel
    snn_chunk_kernel<false><<<dim3(kNCH, kNT), dim3(32), 0, stream>>>(
        x, Wenc, benc, Wmem, bmem, Wdec, bdec,
        nullptr, E, nullptr, nullptr, nullptr);
    // Phase B: A^64 + scan over chunk boundaries
    snn_scanfix_kernel<<<dim3(kNT), dim3(32), 0, stream>>>(Wmem, E, Minit);
    // Phase C: seeded chunks, emit z / recon / zsum partials
    snn_chunk_kernel<true><<<dim3(kNCH, kNT), dim3(32), 0, stream>>>(
        x, Wenc, benc, Wmem, bmem, Wdec, bdec,
        Minit, nullptr, recon, zout, Zpart);
    // Phase D: deterministic reduction + logits
    snn_logits_kernel<<<dim3(kNT), dim3(32), 0, stream>>>(Zpart, Wcls, bcls, logits);
  } else {
    snn_fused_kernel<<<dim3(kNT), dim3(32), 0, stream>>>(
        x, Wenc, benc, Wmem, bmem, Wdec, bdec, Wcls, bcls, recon, logits, zout);
  }
}
